// Mlp_70798240907434
// MI455X (gfx1250) — matrix operand order repaired and checked by reading, not compiled, not run on hardware
//
#include <hip/hip_runtime.h>
#include <hip/hip_bf16.h>
#include <math.h>

typedef __attribute__((ext_vector_type(16))) __bf16 v16bf;
typedef __attribute__((ext_vector_type(8)))  __bf16 v8bf;
typedef __attribute__((ext_vector_type(8)))  float  v8f;

union BFrag {
    v16bf v;
    v8bf  h[2];
};

#define WMMA_BF16(a, b, c) \
    __builtin_amdgcn_wmma_f32_16x16x32_bf16(false, (a), false, (b), (short)0, (c), false, false)

#define NB   4
#define CIN  64
#define CHID 256
#define COUT 64
#define HW   96
#define LTOT 9216
#define PW   98

// ---------------------------------------------------------------------------
// Async global->LDS copy (CDNA5 GLOBAL_LOAD_ASYNC_TO_LDS_B128) via inline asm.
// The clang builtin exists but demands LangAS-qualified pointer types that
// HIP source cannot spell, so we emit the instruction directly. The LDS byte
// offset is the low 32 bits of the generic pointer to a __shared__ object.
// Completion is tracked by ASYNCcnt.
// ---------------------------------------------------------------------------
#define USE_ASYNC_COPY 1

#if defined(__has_builtin)
#if __has_builtin(__builtin_amdgcn_s_wait_asynccnt)
#define HAVE_WAIT_ASYNC_BUILTIN 1
#endif
#endif

__device__ __forceinline__ void cp16(const __bf16* g, __bf16* l) {
#ifdef USE_ASYNC_COPY
    asm volatile("global_load_async_to_lds_b128 %0, %1, off"
                 :: "v"((uint32_t)(uintptr_t)l),
                    "v"((uint64_t)(uintptr_t)g)
                 : "memory");
#else
    *(v8bf*)l = *(const v8bf*)g;      // global_load_b128 + ds_store_b128
#endif
}

__device__ __forceinline__ void wait_async_lds() {
#ifdef USE_ASYNC_COPY
#ifdef HAVE_WAIT_ASYNC_BUILTIN
    __builtin_amdgcn_s_wait_asynccnt(0);
#else
    asm volatile("s_wait_asynccnt 0" ::: "memory");
#endif
#endif
}

// ---------------------------------------------------------------------------
// Fragment loaders (16-bit WMMA operands; this code's contiguous-K order, same for A and B):
//   A (MxK): lane row = lane&15; 16 contiguous K at K = kh*16
//   B (KxN): lane col = lane&15; 16 contiguous K at K = kh*16
// ---------------------------------------------------------------------------
__device__ __forceinline__ BFrag load_a_g(const __bf16* row, int khalf) {
    BFrag f;
    f.h[0] = *(const v8bf*)(row + khalf * 16);
    f.h[1] = *(const v8bf*)(row + khalf * 16 + 8);
    return f;
}
__device__ __forceinline__ BFrag load_b_g(const __bf16* col, int khalf) {
    BFrag f;
    f.h[0] = *(const v8bf*)(col + khalf * 16);
    f.h[1] = *(const v8bf*)(col + khalf * 16 + 8);
    return f;
}

// ---------------------------------------------------------------------------
// Kernel 1: transpose x (b,c,l) f32 -> Xt (b,l,c) bf16  (channels-last)
// ---------------------------------------------------------------------------
__global__ __launch_bounds__(256) void k_xt(const float* __restrict__ x,
                                            __bf16* __restrict__ Xt) {
    int idx = blockIdx.x * 256 + threadIdx.x;
    int c = idx & (CIN - 1);
    int rem = idx >> 6;
    int l = rem % LTOT;
    int b = rem / LTOT;
    Xt[idx] = (__bf16)x[((size_t)b * CIN + c) * LTOT + l];
}

// ---------------------------------------------------------------------------
// Kernel 2: layer-1 gates + modulated weight W1m[b][o][c] (bf16)
// ---------------------------------------------------------------------------
__global__ __launch_bounds__(256) void k_gates1(const float* __restrict__ x,
                                                const float* __restrict__ ce,
                                                const float* __restrict__ gd,
                                                const float* __restrict__ gd2,
                                                const float* __restrict__ ci,
                                                const float* __restrict__ w1,
                                                __bf16* __restrict__ W1m) {
    int b = blockIdx.x;
    int t = threadIdx.x;
    __shared__ float smax[256];
    __shared__ float s_t[CIN];
    __shared__ float colg[CIN];
    __shared__ float rowg[CHID];

    int c = t & (CIN - 1), part = t >> 6;
    const float* xp = x + ((size_t)b * CIN + c) * LTOT + part * (LTOT / 4);
    float m = -3.4e38f;
    for (int i = 0; i < LTOT / 4; ++i) m = fmaxf(m, xp[i]);
    smax[t] = m;
    __syncthreads();

    if (t < CIN) {
        float gm = fmaxf(fmaxf(smax[t], smax[t + 64]),
                         fmaxf(smax[t + 128], smax[t + 192]));
        float ce2v = gm * ce[0];
        float r = fmaxf(ce2v, 0.f);
        s_t[t] = r;
        colg[t] = r * gd[0];
    }
    __syncthreads();

    {
        int o = t, p = o >> 5, oi = o & 31;
        float acc = 0.f;
        #pragma unroll
        for (int g = 0; g < 8; ++g) acc += s_t[p * 8 + g] * ci[oi * 8 + g];
        rowg[o] = fmaxf(acc, 0.f) * gd2[0];
    }
    __syncthreads();

    for (int i = t; i < CHID * CIN; i += 256) {
        int o = i >> 6, cc = i & (CIN - 1);
        float g = 1.f / (1.f + expf(-(colg[cc] + rowg[o])));
        W1m[(size_t)b * CHID * CIN + i] = (__bf16)(g * w1[i]);
    }
}

// ---------------------------------------------------------------------------
// Kernel 3: layer-1 GEMM  D[l][o] = sum_c Xt[l][c]*W1m[o][c], GELU -> Hp bf16.
//   A = Xt (M=spatial rows, global), B = W1m (N=oc cols, staged in LDS).
//   LDS swizzle: W1m row o has 8 16B-chunks; chunk c stored at c^(o&7).
// ---------------------------------------------------------------------------
__global__ __launch_bounds__(256) void k_l1(const __bf16* __restrict__ Xt,
                                            const __bf16* __restrict__ W1m,
                                            __bf16* __restrict__ Hp) {
    __shared__ __bf16 sW[CHID * CIN];   // 32 KB, swizzled
    int b = blockIdx.y;
    int tid = threadIdx.x;
    int wave = tid >> 5, lane = tid & 31;
    int base = blockIdx.x * 128 + wave * 16;
    int khalf = lane >> 4;

    {   // cooperative stage of W1m[b] (2048 chunks of 16B, 8 per thread)
        const __bf16* gsrc = W1m + (size_t)b * CHID * CIN;
        #pragma unroll
        for (int r = 0; r < 8; ++r) {
            int lin = r * 256 + tid;
            int o = lin >> 3, cch = lin & 7;
            int dst = (o << 3) | (cch ^ (o & 7));
            cp16(gsrc + (size_t)lin * 8, sW + (size_t)dst * 8);
        }
    }
    wait_async_lds();
    __syncthreads();

    const __bf16* A = Xt + ((size_t)(b * LTOT + base + (lane & 15))) * CIN;
    BFrag a0 = load_a_g(A, khalf);          // K 0..31
    BFrag a1 = load_a_g(A + 32, khalf);     // K 32..63

    for (int n0 = 0; n0 < CHID; n0 += 64) {
        v8f acc[4];
        #pragma unroll
        for (int t = 0; t < 4; ++t) acc[t] = (v8f){0.f,0.f,0.f,0.f,0.f,0.f,0.f,0.f};

        #pragma unroll
        for (int t = 0; t < 4; ++t) {
            int o = n0 + t * 16 + (lane & 15);
            int sw = o & 7;
            BFrag f0, f1;
            int c0 = 2 * khalf;             // kc = 0
            f0.h[0] = *(const v8bf*)(sW + (((o << 3) | (c0 ^ sw)) << 3));
            f0.h[1] = *(const v8bf*)(sW + (((o << 3) | ((c0 + 1) ^ sw)) << 3));
            int c1 = 4 + 2 * khalf;         // kc = 32
            f1.h[0] = *(const v8bf*)(sW + (((o << 3) | (c1 ^ sw)) << 3));
            f1.h[1] = *(const v8bf*)(sW + (((o << 3) | ((c1 + 1) ^ sw)) << 3));
            acc[t] = WMMA_BF16(a0.v, f0.v, acc[t]);
            acc[t] = WMMA_BF16(a1.v, f1.v, acc[t]);
        }

        #pragma unroll
        for (int t = 0; t < 4; ++t) {
            int o = n0 + t * 16 + (lane & 15);
            #pragma unroll
            for (int j = 0; j < 8; ++j) {
                int l = base + j + khalf * 8;
                float v = acc[t][j];
                v = 0.5f * v * (1.f + erff(v * 0.70710678118f));
                int y = l / HW, xx = l % HW;
                Hp[(((size_t)b * PW + y + 1) * PW + (xx + 1)) * CHID + o] = (__bf16)v;
            }
        }
    }
}

// ---------------------------------------------------------------------------
// Kernel 4: 3x3 adaptive max-pool over Hp interior -> gl2[b][c][k] (f32)
// ---------------------------------------------------------------------------
__global__ __launch_bounds__(256) void k_pool2(const __bf16* __restrict__ Hp,
                                               float* __restrict__ gl2) {
    int b = blockIdx.x / 9, k = blockIdx.x % 9;
    int c = threadIdx.x;
    int ky = k / 3, kx = k % 3;
    float m = -3.4e38f;
    for (int y = 0; y < 32; ++y)
        for (int x = 0; x < 32; ++x) {
            float v = (float)Hp[(((size_t)b * PW + ky * 32 + y + 1) * PW
                               + (kx * 32 + x + 1)) * CHID + c];
            m = fmaxf(m, v);
        }
    gl2[((size_t)b * CHID + c) * 9 + k] = m;
}

// ---------------------------------------------------------------------------
// Kernel 5: layer-2 gates + modulated weight W2m[b][dk][o][c] (bf16)
// ---------------------------------------------------------------------------
__global__ __launch_bounds__(256) void k_gates2(const float* __restrict__ gl2,
                                                const float* __restrict__ ce,
                                                const float* __restrict__ gd,
                                                const float* __restrict__ gd2,
                                                const float* __restrict__ ci,
                                                const float* __restrict__ w2,
                                                __bf16* __restrict__ W2m) {
    int b = blockIdx.x;
    int t = threadIdx.x;
    __shared__ float rce[CHID][5];
    __shared__ float colg[CHID][9];
    __shared__ float rowg[COUT][9];

    {
        int c = t;
        float g[9];
        #pragma unroll
        for (int k = 0; k < 9; ++k) g[k] = gl2[((size_t)b * CHID + c) * 9 + k];
        float r[5];
        #pragma unroll
        for (int n = 0; n < 5; ++n) {
            float s = 0.f;
            #pragma unroll
            for (int k = 0; k < 9; ++k) s += g[k] * ce[n * 9 + k];
            r[n] = fmaxf(s, 0.f);
            rce[c][n] = r[n];
        }
        #pragma unroll
        for (int k = 0; k < 9; ++k) {
            float s = 0.f;
            #pragma unroll
            for (int n = 0; n < 5; ++n) s += r[n] * gd[k * 5 + n];
            colg[c][k] = s;
        }
    }
    __syncthreads();

    if (t < COUT) {
        int o = t, p = o >> 1, oi = o & 1;
        float on[5];
        #pragma unroll
        for (int n = 0; n < 5; ++n) {
            float s = 0.f;
            #pragma unroll
            for (int g = 0; g < 8; ++g) s += rce[p * 8 + g][n] * ci[oi * 8 + g];
            on[n] = fmaxf(s, 0.f);
        }
        #pragma unroll
        for (int k = 0; k < 9; ++k) {
            float s = 0.f;
            #pragma unroll
            for (int n = 0; n < 5; ++n) s += on[n] * gd2[k * 5 + n];
            rowg[o][k] = s;
        }
    }
    __syncthreads();

    for (int i = t; i < 9 * COUT * CHID; i += 256) {
        int dk = i / (COUT * CHID);
        int rem = i - dk * COUT * CHID;
        int o = rem >> 8, c = rem & 255;
        float g = 1.f / (1.f + expf(-(colg[c][dk] + rowg[o][dk])));
        W2m[(size_t)b * 9 * COUT * CHID + i] =
            (__bf16)(g * w2[((size_t)o * CHID + c) * 9 + dk]);
    }
}

// ---------------------------------------------------------------------------
// Kernel 6: layer-2 implicit 3x3 conv GEMM, M = oc (64), N = spatial.
//   out[b][o][l] = sum_{dk,c} W2m[b][dk][o][c] * Hp[b][l+off(dk)][c]
//   A = W2m dk-slice staged in LDS (double-buffered, XOR-swizzled chunks),
//   B = Hp rows from global (per-lane spatial column, K-contiguous).
//   D cols = spatial -> coalesced 64B output stores.
// ---------------------------------------------------------------------------
__global__ __launch_bounds__(256) void k_l2(const __bf16* __restrict__ Hp,
                                            const __bf16* __restrict__ W2m,
                                            float* __restrict__ out) {
    __shared__ __bf16 sW[2][COUT * CHID];   // 2 x 32 KB, swizzled
    int b = blockIdx.y;
    int tid = threadIdx.x;
    int wave = tid >> 5, lane = tid & 31;
    int base = blockIdx.x * 128 + wave * 16;
    int khalf = lane >> 4;

    int l = base + (lane & 15);
    int y = l / HW, xx = l % HW;
    const __bf16* hRow = Hp + (((size_t)b * PW + y + 1) * PW + (xx + 1)) * CHID;
    const __bf16* Wb = W2m + (size_t)b * 9 * COUT * CHID;

    // stage dk = 0
    #pragma unroll
    for (int r = 0; r < 8; ++r) {
        int lin = r * 256 + tid;
        int o = lin >> 5, cch = lin & 31;
        int dst = (o << 5) | (cch ^ (o & 31));
        cp16(Wb + (size_t)lin * 8, &sW[0][0] + (size_t)dst * 8);
    }
    wait_async_lds();
    __syncthreads();

    v8f acc[4];
    #pragma unroll
    for (int t = 0; t < 4; ++t) acc[t] = (v8f){0.f,0.f,0.f,0.f,0.f,0.f,0.f,0.f};

    for (int dk = 0; dk < 9; ++dk) {
        int cur = dk & 1;
        if (dk < 8) {   // prefetch next slice into the other buffer
            const __bf16* gsrc = Wb + (size_t)(dk + 1) * COUT * CHID;
            #pragma unroll
            for (int r = 0; r < 8; ++r) {
                int lin = r * 256 + tid;
                int o = lin >> 5, cch = lin & 31;
                int dst = (o << 5) | (cch ^ (o & 31));
                cp16(gsrc + (size_t)lin * 8, &sW[1 - cur][0] + (size_t)dst * 8);
            }
        }

        int di = dk / 3 - 1, dj = dk % 3 - 1;
        const __bf16* hB = hRow + (di * PW + dj) * CHID;
        __builtin_prefetch(hB, 0, 1);
        const __bf16* sA = &sW[cur][0];

        #pragma unroll 2
        for (int kc = 0; kc < CHID; kc += 32) {
            BFrag bh = load_b_g(hB + kc, khalf);    // B: 16 contiguous K
            #pragma unroll
            for (int t = 0; t < 4; ++t) {
                int o = t * 16 + (lane & 15);
                int sw = o & 31;
                int cA = (kc >> 3) + 2 * khalf;      // A chunks: cA, cA+1
                BFrag af;
                af.h[0] = *(const v8bf*)(sA + (((o << 5) | (cA ^ sw)) << 3));
                af.h[1] = *(const v8bf*)(sA + (((o << 5) | ((cA + 1) ^ sw)) << 3));
                acc[t] = WMMA_BF16(af.v, bh.v, acc[t]);
            }
        }

        wait_async_lds();
        __syncthreads();
    }

    #pragma unroll
    for (int t = 0; t < 4; ++t) {
        #pragma unroll
        for (int j = 0; j < 8; ++j) {
            int o = t * 16 + j + khalf * 8;
            out[((size_t)b * COUT + o) * LTOT + base + (lane & 15)] = acc[t][j];
        }
    }
}

// ---------------------------------------------------------------------------
extern "C" void kernel_launch(void* const* d_in, const int* in_sizes, int n_in,
                              void* d_out, int out_size, void* d_ws, size_t ws_size,
                              hipStream_t stream) {
    const float* x    = (const float*)d_in[0];
    const float* w1   = (const float*)d_in[1];
    const float* ce1  = (const float*)d_in[2];
    const float* gd1  = (const float*)d_in[3];
    const float* gd21 = (const float*)d_in[4];
    const float* ci1  = (const float*)d_in[5];
    const float* w2   = (const float*)d_in[6];
    const float* ce2  = (const float*)d_in[7];
    const float* gd2  = (const float*)d_in[8];
    const float* gd22 = (const float*)d_in[9];
    const float* ci2  = (const float*)d_in[10];
    float* out = (float*)d_out;

    char* ws = (char*)d_ws;
    size_t off = 0;
    const size_t XT_BYTES  = (size_t)NB * LTOT * CIN * 2;
    const size_t W1M_BYTES = (size_t)NB * CHID * CIN * 2;
    const size_t HP_BYTES  = (size_t)NB * PW * PW * CHID * 2;
    const size_t GL2_BYTES = (size_t)NB * CHID * 9 * 4;
    const size_t W2M_BYTES = (size_t)NB * 9 * COUT * CHID * 2;

    __bf16* Xt  = (__bf16*)(ws + off); off = (off + XT_BYTES  + 255) & ~(size_t)255;
    __bf16* W1m = (__bf16*)(ws + off); off = (off + W1M_BYTES + 255) & ~(size_t)255;
    __bf16* Hp  = (__bf16*)(ws + off); off = (off + HP_BYTES  + 255) & ~(size_t)255;
    float*  gl2 = (float*) (ws + off); off = (off + GL2_BYTES + 255) & ~(size_t)255;
    __bf16* W2m = (__bf16*)(ws + off); off = (off + W2M_BYTES + 255) & ~(size_t)255;

    (void)hipMemsetAsync(Hp, 0, HP_BYTES, stream);

    k_xt    <<<(NB * LTOT * CIN) / 256, 256, 0, stream>>>(x, Xt);
    k_gates1<<<NB, 256, 0, stream>>>(x, ce1, gd1, gd21, ci1, w1, W1m);
    k_l1    <<<dim3(LTOT / 128, NB), 256, 0, stream>>>(Xt, W1m, Hp);
    k_pool2 <<<NB * 9, 256, 0, stream>>>(Hp, gl2);
    k_gates2<<<NB, 256, 0, stream>>>(gl2, ce2, gd2, gd22, ci2, w2, W2m);
    k_l2    <<<dim3(LTOT / 128, NB), 256, 0, stream>>>(Hp, W2m, out);
}
